// customLSTMcell_81406810129161
// MI455X (gfx1250) — matrix operand order repaired and checked by reading, not compiled, not run on hardware
//
#include <hip/hip_runtime.h>

#define B_  128
#define T_  512
#define NI_ 512
#define N_  512

typedef __attribute__((ext_vector_type(16))) __bf16 v16bf;
typedef __attribute__((ext_vector_type(8)))  float  v8f;

union Frag { uint4 q[2]; v16bf v; };

#if __has_builtin(__builtin_amdgcn_global_load_async_to_lds_b128)
#define HAVE_ASYNC 1
typedef int v4i __attribute__((vector_size(16)));
typedef __attribute__((address_space(3))) v4i lv4i;
#else
#define HAVE_ASYNC 0
#endif

static __device__ __forceinline__ unsigned short f2bf(float f) {
  unsigned int u = __float_as_uint(f);
  unsigned int r = u + 0x7fffu + ((u >> 16) & 1u);   // round-to-nearest-even
  return (unsigned short)(r >> 16);
}

// ---------------- Kernel 1: x (f32) -> xbf (bf16), same [B,T,NI] layout ----
__global__ void conv_x_kernel(const float* __restrict__ x,
                              unsigned short* __restrict__ xbf, long long n) {
  long long i4 = ((long long)blockIdx.x * blockDim.x + threadIdx.x) * 4;
  if (i4 >= n) return;
  float4 f = *(const float4*)(x + i4);
  uint2 o;
  o.x = (unsigned int)f2bf(f.x) | ((unsigned int)f2bf(f.y) << 16);
  o.y = (unsigned int)f2bf(f.z) | ((unsigned int)f2bf(f.w) << 16);
  *(uint2*)(xbf + i4) = o;
}

// ---------------- Kernel 2: pack W/U into WMMA B-fragment layout -----------
// Packed index (dwords): ((tt*16 + kk)*32 + lane)*8 + v
//   tt = gate*32 + n_tile (16 cols), kk = K-chunk of 32, lane per B layout:
//   lanes 0-15: n = lane, K = kchunk*32 + 16*(v/4) + 2*(v%4) + {0,1}
//   lanes16-31: n = lane-16, K = kchunk*32 + 8 + 16*(v/4) + 2*(v%4) + {0,1}
__global__ void pack_wu_kernel(const float* __restrict__ wf, const float* __restrict__ wi,
                               const float* __restrict__ wo, const float* __restrict__ wc,
                               const float* __restrict__ uf, const float* __restrict__ ui,
                               const float* __restrict__ uo, const float* __restrict__ uc,
                               unsigned int* __restrict__ Wpk, unsigned int* __restrict__ Upk) {
  const unsigned int PER = 128u * 16u * 32u * 8u;   // 524288 dwords per matrix
  unsigned int idx = blockIdx.x * blockDim.x + threadIdx.x;
  if (idx >= 2u * PER) return;
  bool isU = idx >= PER;
  unsigned int d = isU ? idx - PER : idx;
  unsigned int widx = d;
  unsigned int v = d & 7u;   d >>= 3;
  unsigned int lane = d & 31u; d >>= 5;
  unsigned int kk = d & 15u; d >>= 4;
  unsigned int tt = d;                       // 0..127
  unsigned int gate = tt >> 5;
  unsigned int n  = ((tt & 31u) << 4) + (lane & 15u);
  unsigned int k0 = kk * 32u + ((v >> 2) << 4) + ((lane >> 4) << 3) + 2u * (v & 3u);
  const float* m;
  if (isU) m = (gate == 0) ? uf : (gate == 1) ? ui : (gate == 2) ? uo : uc;
  else     m = (gate == 0) ? wf : (gate == 1) ? wi : (gate == 2) ? wo : wc;
  float s0 = m[(size_t)k0 * N_ + n];
  float s1 = m[(size_t)(k0 + 1) * N_ + n];
  unsigned int val = (unsigned int)f2bf(s0) | ((unsigned int)f2bf(s1) << 16);
  (isU ? Upk : Wpk)[widx] = val;
}

// ---------------- Kernel 3: persistent fused LSTM scan ---------------------
// 8 blocks x 1024 threads (32 waves). Block owns batch rows [b0,b0+16).
// Wave w owns hidden columns [16w,16w+16) and all 4 gates for them.
__launch_bounds__(1024, 1)
__global__ void lstm_scan_kernel(const unsigned short* __restrict__ xbf,
                                 const unsigned int* __restrict__ Wpk,
                                 const unsigned int* __restrict__ Upk,
                                 const float* __restrict__ bfv, const float* __restrict__ biv,
                                 const float* __restrict__ bov, const float* __restrict__ bcv,
                                 const float* __restrict__ c0, const float* __restrict__ h0,
                                 float* __restrict__ out) {
  float* out_hs = out;
  float* out_c  = out + (size_t)B_ * T_ * N_;
  float* out_h  = out_c + (size_t)B_ * N_;

  const int tid  = threadIdx.x;
  const int wave = tid >> 5;
  const int lane = tid & 31;
  const int b0   = blockIdx.x * 16;
  const int n0   = wave * 16;
  const int hl   = lane >> 4;      // lane half selects K sub-range (A) / M+8 (C)
  const int nn   = lane & 15;
  const int mrow = hl * 8;         // C/D layout: reg r -> row M = mrow + r

  __shared__ unsigned short hbuf[16][N_ + 8];
  __shared__ unsigned short xbuf[2][16][NI_ + 8];

  for (int i = tid; i < 16 * N_; i += 1024) {
    int m = i >> 9, n = i & (N_ - 1);
    hbuf[m][n] = f2bf(h0[(size_t)(b0 + m) * N_ + n]);
  }

  float creg[8];
#pragma unroll
  for (int r = 0; r < 8; ++r)
    creg[r] = c0[(size_t)(b0 + mrow + r) * N_ + n0 + nn];

  float bias[4];
  bias[0] = bfv[n0 + nn]; bias[1] = biv[n0 + nn];
  bias[2] = bov[n0 + nn]; bias[3] = bcv[n0 + nn];

  const uint4* wb[4]; const uint4* ub[4];
#pragma unroll
  for (int g = 0; g < 4; ++g) {
    size_t base = ((size_t)(g * 32 + wave) * 16 * 32 + lane) * 2;  // uint4 units
    wb[g] = (const uint4*)Wpk + base;
    ub[g] = (const uint4*)Upk + base;
  }

  const int xrow = tid >> 6;       // 16 rows x 64 chunks of 16B = one x tile
  const int xchunk = tid & 63;
  auto prefetch = [&](int buf, int t) {
    const unsigned short* gsrc =
        xbf + ((size_t)(b0 + xrow) * T_ + t) * NI_ + xchunk * 8;
    unsigned short* ldst = &xbuf[buf][xrow][xchunk * 8];
#if HAVE_ASYNC
    __builtin_amdgcn_global_load_async_to_lds_b128((v4i*)gsrc, (lv4i*)ldst, 0, 0);
#else
    *(uint4*)ldst = *(const uint4*)gsrc;
#endif
  };

  prefetch(0, 0);
  __syncthreads();

  for (int t = 0; t < T_; ++t) {
    const int buf = t & 1;
    if (t + 1 < T_) {
      prefetch(buf ^ 1, t + 1);
      asm volatile("s_wait_asynccnt 1" ::: "memory");
    } else {
      asm volatile("s_wait_asynccnt 0" ::: "memory");
    }
    __syncthreads();   // x[t] staged by all waves; hbuf(t) writes visible

    v8f acc[4];
#pragma unroll
    for (int g = 0; g < 4; ++g) {
      v8f a;
#pragma unroll
      for (int r = 0; r < 8; ++r) a[r] = bias[g];
      acc[g] = a;
    }

    const unsigned short* hrow = &hbuf[nn][0];
    const unsigned short* xr   = &xbuf[buf][nn][0];
#pragma unroll 2
    for (int kk = 0; kk < 16; ++kk) {
      Frag ah, ax;
      const int k0 = kk * 32 + hl * 8;     // 16-bit A layout (ISA 7.12.2)
      ah.q[0] = *(const uint4*)(hrow + k0);
      ah.q[1] = *(const uint4*)(hrow + k0 + 16);
      ax.q[0] = *(const uint4*)(xr + k0);
      ax.q[1] = *(const uint4*)(xr + k0 + 16);
#pragma unroll
      for (int g = 0; g < 4; ++g) {
        Frag bu, bw;
        bu.q[0] = ub[g][kk * 64 + 0];
        bu.q[1] = ub[g][kk * 64 + 1];
        bw.q[0] = wb[g][kk * 64 + 0];
        bw.q[1] = wb[g][kk * 64 + 1];
        acc[g] = __builtin_amdgcn_wmma_f32_16x16x32_bf16(
            false, ah.v, false, bu.v, (short)0, acc[g], false, false);
        acc[g] = __builtin_amdgcn_wmma_f32_16x16x32_bf16(
            false, ax.v, false, bw.v, (short)0, acc[g], false, false);
      }
    }

    __syncthreads();   // all waves finished reading hbuf(t)

    float hv[8];
#pragma unroll
    for (int r = 0; r < 8; ++r) {
      float fg = 1.f / (1.f + __expf(-acc[0][r]));
      float ig = 1.f / (1.f + __expf(-acc[1][r]));
      float og = 1.f / (1.f + __expf(-acc[2][r]));
      float gg = tanhf(acc[3][r]);
      float c  = fg * creg[r] + ig * gg;
      creg[r]  = c;
      hv[r]    = og * tanhf(c);
    }
#pragma unroll
    for (int r = 0; r < 8; ++r) {
      const int M = mrow + r;
      hbuf[M][n0 + nn] = f2bf(hv[r]);
      out_hs[((size_t)(b0 + M) * T_ + t) * N_ + n0 + nn] = hv[r];
    }
    if (t == T_ - 1) {
#pragma unroll
      for (int r = 0; r < 8; ++r) {
        const int M = mrow + r;
        out_c[(size_t)(b0 + M) * N_ + n0 + nn] = creg[r];
        out_h[(size_t)(b0 + M) * N_ + n0 + nn] = hv[r];
      }
    }
  }
}

extern "C" void kernel_launch(void* const* d_in, const int* in_sizes, int n_in,
                              void* d_out, int out_size, void* d_ws, size_t ws_size,
                              hipStream_t stream) {
  const float* x  = (const float*)d_in[0];
  const float* c0 = (const float*)d_in[1];
  const float* h0 = (const float*)d_in[2];
  const float* wf = (const float*)d_in[3];
  const float* uf = (const float*)d_in[4];
  const float* bf = (const float*)d_in[5];
  const float* wi = (const float*)d_in[6];
  const float* ui = (const float*)d_in[7];
  const float* bi = (const float*)d_in[8];
  const float* wo = (const float*)d_in[9];
  const float* uo = (const float*)d_in[10];
  const float* bo = (const float*)d_in[11];
  const float* wc = (const float*)d_in[12];
  const float* uc = (const float*)d_in[13];
  const float* bc = (const float*)d_in[14];

  unsigned short* xbf = (unsigned short*)d_ws;                        // 64 MB bf16
  unsigned int* Wpk = (unsigned int*)((char*)d_ws + (size_t)B_ * T_ * NI_ * 2);
  unsigned int* Upk = Wpk + 524288;                                   // 2 MB each

  const long long nx = (long long)B_ * T_ * NI_;
  conv_x_kernel<<<(unsigned)(nx / 4 / 256), 256, 0, stream>>>(x, xbf, nx);
  pack_wu_kernel<<<(2 * 524288) / 256, 256, 0, stream>>>(wf, wi, wo, wc,
                                                         uf, ui, uo, uc, Wpk, Upk);
  lstm_scan_kernel<<<B_ / 16, 1024, 0, stream>>>(xbf, Wpk, Upk,
                                                 bf, bi, bo, bc, c0, h0,
                                                 (float*)d_out);
}
